// SlotAttentionWithPositions_5815385719116
// MI455X (gfx1250) — matrix operand order repaired and checked by reading, not compiled, not run on hardware
//
#include <hip/hip_runtime.h>
#include <hip/hip_bf16.h>
#include <math.h>

typedef _Float16 h16;
typedef __attribute__((ext_vector_type(16))) _Float16 v16h;
typedef __attribute__((ext_vector_type(8)))  _Float16 v8h;
typedef __attribute__((ext_vector_type(8)))  float    v8f;

#define B_      128
#define EMB_    128
#define KD_     128
#define RES_    32
#define NTOK_   1024
#define NSLOTS_ 7
#define SLOTP_  16          // slots padded to 16 rows for WMMA
#define MTOT_   (B_*NTOK_)  // 131072
#define SCALE_  0.08838834764831843f
#define EPS_    1e-8f

#define WSTRIDE_ 136                    // padded K-stride of the LDS weight image
#define WIMG_ELEMS_ (128 * WSTRIDE_)    // 17408 halves = 34816 bytes per matrix
#define WIMG_CHUNKS_ (WIMG_ELEMS_ * 2 / 16)  // 2176 x 16B async transfers

__device__ __forceinline__ float geluf(float x) {
    return 0.5f * x * (1.0f + erff(x * 0.7071067811865476f));
}
__device__ __forceinline__ float sigm(float x) { return 1.0f / (1.0f + expf(-x)); }

// ---------------------------------------------------------------------------
// Encoder: tokens = transpose(inputs + pos), LayerNorm over [NTOK,EMB] per b,
// write f16 tokens. One block per batch sample; e-major loop => coalesced reads.
// ---------------------------------------------------------------------------
__global__ __launch_bounds__(256)
void enc_build_ln_kernel(const float* __restrict__ inp,
                         const float* __restrict__ pw, const float* __restrict__ pb,
                         const float* __restrict__ g,  const float* __restrict__ bb,
                         h16* __restrict__ out)
{
    __shared__ float s_pw[4 * EMB_];
    __shared__ float s_pb[EMB_];
    __shared__ float red[256];
    int b = blockIdx.x, tid = threadIdx.x;
    for (int i = tid; i < 4 * EMB_; i += 256) s_pw[i] = pw[i];
    for (int i = tid; i < EMB_; i += 256) s_pb[i] = pb[i];
    __syncthreads();
    const float* ib = inp + (size_t)b * EMB_ * NTOK_;
    float sum = 0.f, sq = 0.f;
    for (int idx = tid; idx < EMB_ * NTOK_; idx += 256) {
        int e = idx >> 10, t = idx & 1023;
        int i = t >> 5, j = t & 31;
        float xx = i * (1.0f / 31.0f), yy = j * (1.0f / 31.0f);
        float p = xx * s_pw[e] + yy * s_pw[EMB_ + e] +
                  (1.f - xx) * s_pw[2 * EMB_ + e] + (1.f - yy) * s_pw[3 * EMB_ + e] + s_pb[e];
        float x = ib[idx] + p;
        sum += x; sq += x * x;
    }
    red[tid] = sum; __syncthreads();
    for (int o = 128; o > 0; o >>= 1) { if (tid < o) red[tid] += red[tid + o]; __syncthreads(); }
    float mean = red[0] * (1.0f / (EMB_ * NTOK_));
    __syncthreads();
    red[tid] = sq; __syncthreads();
    for (int o = 128; o > 0; o >>= 1) { if (tid < o) red[tid] += red[tid + o]; __syncthreads(); }
    float var = red[0] * (1.0f / (EMB_ * NTOK_)) - mean * mean;
    float inv = rsqrtf(var + 1e-5f);
    __syncthreads();
    h16* ob = out + (size_t)b * NTOK_ * EMB_;
    for (int idx = tid; idx < EMB_ * NTOK_; idx += 256) {
        int e = idx >> 10, t = idx & 1023;
        int i = t >> 5, j = t & 31;
        float xx = i * (1.0f / 31.0f), yy = j * (1.0f / 31.0f);
        float p = xx * s_pw[e] + yy * s_pw[EMB_ + e] +
                  (1.f - xx) * s_pw[2 * EMB_ + e] + (1.f - yy) * s_pw[3 * EMB_ + e] + s_pb[e];
        float x = ib[idx] + p;
        float y = (x - mean) * inv * g[t * EMB_ + e] + bb[t * EMB_ + e];
        ob[(size_t)t * EMB_ + e] = (h16)y;
    }
}

// ---------------------------------------------------------------------------
// One-time weight prep: convert four 128x128 f32 weights into the exact
// padded, transposed (N-major) f16 LDS image layout used by the GEMM kernel,
// so blocks can async-DMA them straight into LDS.
// ---------------------------------------------------------------------------
__global__ void prep_weights_kernel(const float* __restrict__ w1, const float* __restrict__ w2,
                                    const float* __restrict__ wkp, const float* __restrict__ wvp,
                                    h16* __restrict__ img)
{
    int idx = blockIdx.x * 256 + threadIdx.x;
    if (idx >= 4 * WIMG_ELEMS_) return;
    int m = idx / WIMG_ELEMS_;
    int r = idx - m * WIMG_ELEMS_;
    int n = r / WSTRIDE_, k = r - n * WSTRIDE_;
    const float* w = (m == 0) ? w1 : (m == 1) ? w2 : (m == 2) ? wkp : wvp;
    float v = (k < 128) ? w[k * 128 + n] : 0.f;
    img[idx] = (h16)v;
}

// ---------------------------------------------------------------------------
// WMMA GEMM: C[M,128] = act(A[M,128] @ W[128,128] + bias). A f16 in global,
// W pre-imaged as transposed f16; staged into LDS with async-to-LDS DMA
// (gfx1250 ASYNCcnt path). One 16-row M tile per wave, full N=128 per wave.
// OUTMODE 0: f16 row-major. OUTMODE 1: f16 transposed per-batch [b][n][tok].
// ---------------------------------------------------------------------------
template <int ACT, int OUTMODE>
__global__ __launch_bounds__(256)
void gemm128_kernel(const h16* __restrict__ A, const h16* __restrict__ wimg,
                    const float* __restrict__ bias, h16* __restrict__ outp)
{
    __shared__ __attribute__((aligned(16))) h16 wt[WIMG_ELEMS_];
    int tid = threadIdx.x;

    // async DMA the 34,816-byte weight image into LDS (16B per lane per op)
    unsigned ldsbase = (unsigned)(uintptr_t)(&wt[0]);
    for (int c = tid; c < WIMG_CHUNKS_; c += 256) {
        unsigned loff = ldsbase + (unsigned)c * 16u;
        unsigned goff = (unsigned)c * 16u;
        asm volatile("global_load_async_to_lds_b128 %0, %1, %2"
                     :: "v"(loff), "v"(goff), "s"(wimg)
                     : "memory");
    }
    asm volatile("s_wait_asynccnt 0x0" ::: "memory");
    __syncthreads();

    int wave = tid >> 5, lane = tid & 31;
    int mrow = lane & 15, hi = lane >> 4;
    int m0 = blockIdx.x * 128 + wave * 16;
    const h16* Arow = A + (size_t)(m0 + mrow) * 128;
    v8f acc[8];
#pragma unroll
    for (int nt = 0; nt < 8; ++nt)
#pragma unroll
        for (int r = 0; r < 8; ++r) acc[nt][r] = 0.0f;

#pragma unroll
    for (int kc = 0; kc < 4; ++kc) {
        v8h alo = *reinterpret_cast<const v8h*>(Arow + kc * 32 + hi * 8);
        v8h ahi = *reinterpret_cast<const v8h*>(Arow + kc * 32 + 16 + hi * 8);
        v16h af;
#pragma unroll
        for (int i = 0; i < 8; ++i) { af[i] = alo[i]; af[i + 8] = ahi[i]; }
#pragma unroll
        for (int nt = 0; nt < 8; ++nt) {
            const h16* bp = &wt[(nt * 16 + mrow) * WSTRIDE_ + kc * 32 + hi * 8];
            v8h blo = *reinterpret_cast<const v8h*>(bp);
            v8h bhi = *reinterpret_cast<const v8h*>(bp + 16);
            v16h bf;
#pragma unroll
            for (int i = 0; i < 8; ++i) { bf[i] = blo[i]; bf[i + 8] = bhi[i]; }
            acc[nt] = __builtin_amdgcn_wmma_f32_16x16x32_f16(
                false, af, false, bf, (short)0, acc[nt], false, false);
        }
    }
#pragma unroll
    for (int nt = 0; nt < 8; ++nt) {
        int n = nt * 16 + mrow;
        float bv = bias[n];
#pragma unroll
        for (int r = 0; r < 8; ++r) {
            int m = m0 + r + 8 * hi;
            float val = acc[nt][r] + bv;
            if (ACT) val = geluf(val);
            if (OUTMODE == 0)
                outp[(size_t)m * 128 + n] = (h16)val;
            else
                outp[(size_t)(m >> 10) * (128 * NTOK_) + (size_t)n * NTOK_ + (m & 1023)] = (h16)val;
        }
    }
}

// ---------------------------------------------------------------------------
// LayerNorm over last dim (128) of f16 tokens -> f16. One wave per token.
// ---------------------------------------------------------------------------
__global__ __launch_bounds__(256)
void ln_lastdim_kernel(const h16* __restrict__ X, const float* __restrict__ g,
                       const float* __restrict__ bb, h16* __restrict__ out)
{
    int tid = threadIdx.x, lane = tid & 31, wave = tid >> 5;
    size_t t = (size_t)blockIdx.x * 8 + wave;
    const h16* xr = X + t * 128;
    float x[4], sum = 0.f, sq = 0.f;
#pragma unroll
    for (int i = 0; i < 4; ++i) { x[i] = (float)xr[lane + i * 32]; sum += x[i]; sq += x[i] * x[i]; }
#pragma unroll
    for (int o = 16; o > 0; o >>= 1) { sum += __shfl_xor(sum, o, 32); sq += __shfl_xor(sq, o, 32); }
    float mean = sum * (1.f / 128.f);
    float var = sq * (1.f / 128.f) - mean * mean;
    float inv = rsqrtf(var + 1e-5f);
    h16* orow = out + t * 128;
#pragma unroll
    for (int i = 0; i < 4; ++i) {
        int d = lane + i * 32;
        orow[d] = (h16)((x[i] - mean) * inv * g[d] + bb[d]);
    }
}

// slots = loc + exp(0.5*logscale)*noise; padded rows zeroed
__global__ void slot_init_kernel(const float* __restrict__ noise, const float* __restrict__ loc,
                                 const float* __restrict__ lsc, float* __restrict__ slots)
{
    int idx = blockIdx.x * 256 + threadIdx.x;
    if (idx >= B_ * SLOTP_ * KD_) return;
    int d = idx & 127, s = (idx >> 7) & 15, b = idx >> 11;
    float v = 0.f;
    if (s < NSLOTS_) v = loc[d] + expf(0.5f * lsc[d]) * noise[(b * NSLOTS_ + s) * KD_ + d];
    slots[idx] = v;
}

__global__ void zero_h16_kernel(h16* p, int n)
{
    int i = blockIdx.x * 256 + threadIdx.x;
    if (i < n) p[i] = (h16)0.f;
}

__global__ void zero_f32_kernel(float* p, size_t n)
{
    size_t i = (size_t)blockIdx.x * 256 + threadIdx.x;
    if (i < n) p[i] = 0.f;
}

// ---------------------------------------------------------------------------
// sp = LN(slots), q = sp@wq+bq (f16, padded rows zero). One block per batch.
// ---------------------------------------------------------------------------
__global__ __launch_bounds__(256)
void slots_norm_q_kernel(const float* __restrict__ slots, const float* __restrict__ g,
                         const float* __restrict__ bb, const float* __restrict__ wq,
                         const float* __restrict__ bq, float* __restrict__ sp,
                         h16* __restrict__ qh)
{
    __shared__ float red1[128], red2[128];
    __shared__ float s_sp[NSLOTS_ * 128];
    int b = blockIdx.x, tid = threadIdx.x;
    for (int s = 0; s < NSLOTS_; ++s) {
        float x = 0.f;
        if (tid < 128) { x = slots[((size_t)b * SLOTP_ + s) * 128 + tid]; red1[tid] = x; red2[tid] = x * x; }
        __syncthreads();
        for (int o = 64; o > 0; o >>= 1) {
            if (tid < o) { red1[tid] += red1[tid + o]; red2[tid] += red2[tid + o]; }
            __syncthreads();
        }
        float mean = red1[0] * (1.f / 128.f);
        float var = red2[0] * (1.f / 128.f) - mean * mean;
        float inv = rsqrtf(var + 1e-5f);
        if (tid < 128) {
            float y = (x - mean) * inv * g[tid] + bb[tid];
            s_sp[s * 128 + tid] = y;
            sp[((size_t)b * SLOTP_ + s) * 128 + tid] = y;
        }
        __syncthreads();
    }
    for (int i = tid; i < (SLOTP_ - NSLOTS_) * 128; i += 256)
        sp[((size_t)b * SLOTP_ + NSLOTS_) * 128 + i] = 0.f;
    for (int idx = tid; idx < SLOTP_ * 128; idx += 256) {
        int s = idx >> 7, n = idx & 127;
        float acc = 0.f;
        if (s < NSLOTS_) {
            acc = bq[n];
            for (int d = 0; d < 128; ++d) acc += s_sp[s * 128 + d] * wq[d * 128 + n];
        }
        qh[(size_t)b * SLOTP_ * 128 + idx] = (h16)acc;
    }
}

// ---------------------------------------------------------------------------
// dots = q @ k^T * SCALE per batch. A = q (16x128 padded), B^T-layout == k
// row-major, so B-frags load straight from k. One block per batch, 8 waves,
// each wave covers 8 token tiles (8 x 16 = 128 tokens).
// ---------------------------------------------------------------------------
__global__ __launch_bounds__(256)
void dots_wmma_kernel(const h16* __restrict__ qh, const h16* __restrict__ kh,
                      float* __restrict__ dots)
{
    int b = blockIdx.x, tid = threadIdx.x;
    int wave = tid >> 5, lane = tid & 31, mrow = lane & 15, hi = lane >> 4;
    const h16* Arow = qh + ((size_t)b * SLOTP_ + mrow) * 128;
    const h16* Kb = kh + (size_t)b * NTOK_ * 128;
    v8f acc[8];
#pragma unroll
    for (int i = 0; i < 8; ++i)
#pragma unroll
        for (int r = 0; r < 8; ++r) acc[i][r] = 0.0f;

#pragma unroll
    for (int kc = 0; kc < 4; ++kc) {
        v8h alo = *reinterpret_cast<const v8h*>(Arow + kc * 32 + hi * 8);
        v8h ahi = *reinterpret_cast<const v8h*>(Arow + kc * 32 + 16 + hi * 8);
        v16h af;
#pragma unroll
        for (int i = 0; i < 8; ++i) { af[i] = alo[i]; af[i + 8] = ahi[i]; }
#pragma unroll
        for (int i = 0; i < 8; ++i) {
            int ntile = wave * 8 + i;
            const h16* bp = Kb + (size_t)(ntile * 16 + mrow) * 128 + kc * 32 + hi * 8;
            v8h blo = *reinterpret_cast<const v8h*>(bp);
            v8h bhi = *reinterpret_cast<const v8h*>(bp + 16);
            v16h bf;
#pragma unroll
            for (int q = 0; q < 8; ++q) { bf[q] = blo[q]; bf[q + 8] = bhi[q]; }
            acc[i] = __builtin_amdgcn_wmma_f32_16x16x32_f16(
                false, af, false, bf, (short)0, acc[i], false, false);
        }
    }
    float* drow = dots + (size_t)b * SLOTP_ * NTOK_;
#pragma unroll
    for (int i = 0; i < 8; ++i) {
        int tcol = (wave * 8 + i) * 16 + mrow;
#pragma unroll
        for (int r = 0; r < 8; ++r) {
            int m = r + 8 * hi;
            drow[(size_t)m * NTOK_ + tcol] = acc[i][r] * SCALE_;
        }
    }
}

// ---------------------------------------------------------------------------
// Softmax over slot axis per token (+EPS) -> attn_vis (d_out), then renorm
// per slot over tokens -> attn f16 (padded rows stay zero). Block per batch.
// ---------------------------------------------------------------------------
__global__ __launch_bounds__(256)
void softmax_kernel(float* __restrict__ dots, h16* __restrict__ attn,
                    float* __restrict__ attn_vis)
{
    __shared__ float red[256];
    __shared__ float rowsum[NSLOTS_];
    int b = blockIdx.x, tid = threadIdx.x;
    float* dp = dots + (size_t)b * SLOTP_ * NTOK_;
    float ls[NSLOTS_];
#pragma unroll
    for (int s = 0; s < NSLOTS_; ++s) ls[s] = 0.f;
    for (int t = tid; t < NTOK_; t += 256) {
        float v[NSLOTS_], mx = -1e30f;
#pragma unroll
        for (int s = 0; s < NSLOTS_; ++s) { v[s] = dp[s * NTOK_ + t]; mx = fmaxf(mx, v[s]); }
        float tot = 0.f;
#pragma unroll
        for (int s = 0; s < NSLOTS_; ++s) { v[s] = expf(v[s] - mx); tot += v[s]; }
        float rt = 1.f / tot;
#pragma unroll
        for (int s = 0; s < NSLOTS_; ++s) {
            float vis = v[s] * rt + EPS_;
            dp[s * NTOK_ + t] = vis;
            attn_vis[((size_t)b * NSLOTS_ + s) * NTOK_ + t] = vis;
            ls[s] += vis;
        }
    }
    for (int s = 0; s < NSLOTS_; ++s) {
        red[tid] = ls[s]; __syncthreads();
        for (int o = 128; o > 0; o >>= 1) { if (tid < o) red[tid] += red[tid + o]; __syncthreads(); }
        if (tid == 0) rowsum[s] = red[0];
        __syncthreads();
    }
    h16* ap = attn + (size_t)b * SLOTP_ * NTOK_;
    for (int t = tid; t < NTOK_; t += 256) {
#pragma unroll
        for (int s = 0; s < NSLOTS_; ++s)
            ap[s * NTOK_ + t] = (h16)(dp[s * NTOK_ + t] / rowsum[s]);
    }
}

// ---------------------------------------------------------------------------
// upd = attn @ v per batch: M=16 (padded), K=1024, N=128. B operand comes
// from precomputed v^T [b][dim][tok] f16 -> contiguous B-frag loads.
// Two accumulators (even/odd k-chunks) halve the WMMA RAW dependency chain.
// ---------------------------------------------------------------------------
__global__ __launch_bounds__(256)
void upd_wmma_kernel(const h16* __restrict__ attn, const h16* __restrict__ vT,
                     float* __restrict__ upd)
{
    int b = blockIdx.x, tid = threadIdx.x;
    int wave = tid >> 5, lane = tid & 31, mrow = lane & 15, hi = lane >> 4;
    const h16* Arow = attn + ((size_t)b * SLOTP_ + mrow) * NTOK_;
    const h16* Brow = vT + ((size_t)b * 128 + wave * 16 + mrow) * NTOK_;
    v8f acc0, acc1;
#pragma unroll
    for (int r = 0; r < 8; ++r) { acc0[r] = 0.0f; acc1[r] = 0.0f; }
    for (int kc = 0; kc < 32; kc += 2) {
#pragma unroll
        for (int u = 0; u < 2; ++u) {
            int kcu = kc + u;
            v8h alo = *reinterpret_cast<const v8h*>(Arow + kcu * 32 + hi * 8);
            v8h ahi = *reinterpret_cast<const v8h*>(Arow + kcu * 32 + 16 + hi * 8);
            v16h af;
#pragma unroll
            for (int i = 0; i < 8; ++i) { af[i] = alo[i]; af[i + 8] = ahi[i]; }
            v8h blo = *reinterpret_cast<const v8h*>(Brow + kcu * 32 + hi * 8);
            v8h bhi = *reinterpret_cast<const v8h*>(Brow + kcu * 32 + 16 + hi * 8);
            v16h bf;
#pragma unroll
            for (int i = 0; i < 8; ++i) { bf[i] = blo[i]; bf[i + 8] = bhi[i]; }
            if (u == 0)
                acc0 = __builtin_amdgcn_wmma_f32_16x16x32_f16(
                    false, af, false, bf, (short)0, acc0, false, false);
            else
                acc1 = __builtin_amdgcn_wmma_f32_16x16x32_f16(
                    false, af, false, bf, (short)0, acc1, false, false);
        }
    }
#pragma unroll
    for (int r = 0; r < 8; ++r) {
        int m = r + 8 * hi;
        upd[((size_t)b * SLOTP_ + m) * 128 + wave * 16 + mrow] = acc0[r] + acc1[r];
    }
}

// ---------------------------------------------------------------------------
// GRUCell(upd, sp) + preff-LN + residual MLP, per batch per slot (VALU).
// ---------------------------------------------------------------------------
__global__ __launch_bounds__(256)
void gru_mlp_kernel(const float* __restrict__ upd, const float* __restrict__ sp,
                    const float* __restrict__ wih, const float* __restrict__ whh,
                    const float* __restrict__ bih, const float* __restrict__ bhh,
                    const float* __restrict__ pg, const float* __restrict__ pb,
                    const float* __restrict__ w1, const float* __restrict__ b1v,
                    const float* __restrict__ w2, const float* __restrict__ b2v,
                    float* __restrict__ slots)
{
    __shared__ float s_u[128], s_h0[128], s_gi[384], s_gh[384];
    __shared__ float s_new[128], s_hn[128], s_f1[128];
    __shared__ float red1[128], red2[128];
    int b = blockIdx.x, tid = threadIdx.x;
    for (int s = 0; s < NSLOTS_; ++s) {
        if (tid < 128) {
            s_u[tid] = upd[((size_t)b * SLOTP_ + s) * 128 + tid];
            s_h0[tid] = sp[((size_t)b * SLOTP_ + s) * 128 + tid];
        }
        __syncthreads();
        for (int j = tid; j < 384; j += 256) {
            float gi = bih[j], gh = bhh[j];
            const float* wi = wih + (size_t)j * 128;
            const float* wh = whh + (size_t)j * 128;
            for (int d = 0; d < 128; ++d) { gi += s_u[d] * wi[d]; gh += s_h0[d] * wh[d]; }
            s_gi[j] = gi; s_gh[j] = gh;
        }
        __syncthreads();
        if (tid < 128) {
            float r = sigm(s_gi[tid] + s_gh[tid]);
            float z = sigm(s_gi[128 + tid] + s_gh[128 + tid]);
            float nn = tanhf(s_gi[256 + tid] + r * s_gh[256 + tid]);
            float sn = (1.f - z) * nn + z * s_h0[tid];
            s_new[tid] = sn;
            red1[tid] = sn; red2[tid] = sn * sn;
        }
        __syncthreads();
        for (int o = 64; o > 0; o >>= 1) {
            if (tid < o) { red1[tid] += red1[tid + o]; red2[tid] += red2[tid + o]; }
            __syncthreads();
        }
        float mean = red1[0] * (1.f / 128.f);
        float var = red2[0] * (1.f / 128.f) - mean * mean;
        float inv = rsqrtf(var + 1e-5f);
        if (tid < 128) s_hn[tid] = (s_new[tid] - mean) * inv * pg[tid] + pb[tid];
        __syncthreads();
        if (tid < 128) {
            float a = b1v[tid];
            for (int k = 0; k < 128; ++k) a += s_hn[k] * w1[k * 128 + tid];
            s_f1[tid] = geluf(a);
        }
        __syncthreads();
        if (tid < 128) {
            float a = b2v[tid];
            for (int k = 0; k < 128; ++k) a += s_f1[k] * w2[k * 128 + tid];
            slots[((size_t)b * SLOTP_ + s) * 128 + tid] = s_new[tid] + geluf(a);
        }
        __syncthreads();
    }
}

// ---------------------------------------------------------------------------
// Decoder: dec[b*7+s][d][i][j] = slots[b][s][d] + grid4(i,j)@dec_pos_w + bias
// ---------------------------------------------------------------------------
__global__ void decoder_kernel(const float* __restrict__ slots, const float* __restrict__ dw,
                               const float* __restrict__ db, float* __restrict__ out)
{
    size_t idx = (size_t)blockIdx.x * 256 + threadIdx.x;
    if (idx >= (size_t)B_ * NSLOTS_ * KD_ * 64) return;
    int j = idx & 7;
    int i = (idx >> 3) & 7;
    int d = (idx >> 6) & 127;
    int bs = (int)(idx >> 13);
    int s = bs % NSLOTS_, b = bs / NSLOTS_;
    float xx = i * (1.f / 7.f), yy = j * (1.f / 7.f);
    float p = xx * dw[d] + yy * dw[128 + d] + (1.f - xx) * dw[256 + d] + (1.f - yy) * dw[384 + d] + db[d];
    out[idx] = slots[((size_t)b * SLOTP_ + s) * 128 + d] + p;
}

// ---------------------------------------------------------------------------
extern "C" void kernel_launch(void* const* d_in, const int* in_sizes, int n_in,
                              void* d_out, int out_size, void* d_ws, size_t ws_size,
                              hipStream_t stream)
{
    (void)in_sizes; (void)n_in; (void)out_size; (void)ws_size;
    const float* inputs      = (const float*)d_in[0];
    const float* slots_noise = (const float*)d_in[1];
    const float* slots_loc   = (const float*)d_in[2];
    const float* slots_lsc   = (const float*)d_in[3];
    const float* enc_pos_w   = (const float*)d_in[4];
    const float* enc_pos_b   = (const float*)d_in[5];
    const float* enc_norm_g  = (const float*)d_in[6];
    const float* enc_norm_b  = (const float*)d_in[7];
    const float* enc_mlp_w1  = (const float*)d_in[8];
    const float* enc_mlp_b1  = (const float*)d_in[9];
    const float* enc_mlp_w2  = (const float*)d_in[10];
    const float* enc_mlp_b2  = (const float*)d_in[11];
    const float* norm_in_g   = (const float*)d_in[12];
    const float* norm_in_b   = (const float*)d_in[13];
    const float* wq          = (const float*)d_in[14];
    const float* bq          = (const float*)d_in[15];
    const float* wk          = (const float*)d_in[16];
    const float* bk          = (const float*)d_in[17];
    const float* wv          = (const float*)d_in[18];
    const float* bv          = (const float*)d_in[19];
    const float* gru_wih     = (const float*)d_in[20];
    const float* gru_whh     = (const float*)d_in[21];
    const float* gru_bih     = (const float*)d_in[22];
    const float* gru_bhh     = (const float*)d_in[23];
    const float* nslots_g    = (const float*)d_in[24];
    const float* nslots_b    = (const float*)d_in[25];
    const float* npreff_g    = (const float*)d_in[26];
    const float* npreff_b    = (const float*)d_in[27];
    const float* tf_w1       = (const float*)d_in[28];
    const float* tf_b1       = (const float*)d_in[29];
    const float* tf_w2       = (const float*)d_in[30];
    const float* tf_b2       = (const float*)d_in[31];
    const float* dec_pos_w   = (const float*)d_in[32];
    const float* dec_pos_b   = (const float*)d_in[33];

    // workspace carving (3 reusable 33.5MB f16 buffers + attention state)
    const size_t BIG = (size_t)MTOT_ * 128 * sizeof(h16);   // 33,554,432
    char* ws = (char*)d_ws;
    h16*   tok  = (h16*)(ws);                 // tok_ln, later tokens_normed
    h16*   t1   = (h16*)(ws + BIG);           // gelu(mlp1), later k_f16
    h16*   t2   = (h16*)(ws + 2 * BIG);       // gelu(mlp2), later vT_f16
    float* dots = (float*)(ws + 3 * BIG);                       // [B][16][1024] f32
    h16*   attn = (h16*)(ws + 3 * BIG + 8388608);               // [B][16][1024] f16
    h16*   qh   = (h16*)(ws + 3 * BIG + 8388608 + 4194304);     // [B][16][128] f16
    float* slots = (float*)(ws + 3 * BIG + 8388608 + 4194304 + 524288);
    float* sp    = slots + B_ * SLOTP_ * KD_;
    float* upd   = sp + B_ * SLOTP_ * KD_;
    h16*   wimg  = (h16*)(upd + B_ * SLOTP_ * KD_);             // 4 padded W^T f16 images

    float* out_dec  = (float*)d_out;
    float* out_attn = out_dec + (size_t)B_ * NSLOTS_ * KD_ * 64;
    float* out_loss = out_attn + (size_t)B_ * NSLOTS_ * NTOK_;

    // weight prep (once per call) + encoder
    prep_weights_kernel<<<(4 * WIMG_ELEMS_ + 255) / 256, 256, 0, stream>>>(
        enc_mlp_w1, enc_mlp_w2, wk, wv, wimg);
    enc_build_ln_kernel<<<B_, 256, 0, stream>>>(inputs, enc_pos_w, enc_pos_b,
                                                enc_norm_g, enc_norm_b, tok);
    gemm128_kernel<1, 0><<<MTOT_ / 128, 256, 0, stream>>>(tok, wimg + 0 * WIMG_ELEMS_, enc_mlp_b1, t1);
    gemm128_kernel<1, 0><<<MTOT_ / 128, 256, 0, stream>>>(t1, wimg + 1 * WIMG_ELEMS_, enc_mlp_b2, t2);
    ln_lastdim_kernel<<<MTOT_ / 8, 256, 0, stream>>>(t2, norm_in_g, norm_in_b, tok);
    gemm128_kernel<0, 0><<<MTOT_ / 128, 256, 0, stream>>>(tok, wimg + 2 * WIMG_ELEMS_, bk, t1);  // k
    gemm128_kernel<0, 1><<<MTOT_ / 128, 256, 0, stream>>>(tok, wimg + 3 * WIMG_ELEMS_, bv, t2);  // v^T

    // slot init + zero padded attn rows (softmax only writes rows 0..6)
    slot_init_kernel<<<(B_ * SLOTP_ * KD_ + 255) / 256, 256, 0, stream>>>(
        slots_noise, slots_loc, slots_lsc, slots);
    zero_h16_kernel<<<(B_ * SLOTP_ * NTOK_ + 255) / 256, 256, 0, stream>>>(
        attn, B_ * SLOTP_ * NTOK_);

    for (int it = 0; it < 3; ++it) {
        slots_norm_q_kernel<<<B_, 256, 0, stream>>>(slots, nslots_g, nslots_b, wq, bq, sp, qh);
        dots_wmma_kernel<<<B_, 256, 0, stream>>>(qh, t1, dots);
        softmax_kernel<<<B_, 256, 0, stream>>>(dots, attn, out_attn);
        upd_wmma_kernel<<<B_, 256, 0, stream>>>(attn, t2, upd);
        gru_mlp_kernel<<<B_, 256, 0, stream>>>(upd, sp, gru_wih, gru_whh, gru_bih, gru_bhh,
                                               npreff_g, npreff_b, tf_w1, tf_b1, tf_w2, tf_b2,
                                               slots);
    }

    decoder_kernel<<<(int)(((size_t)B_ * NSLOTS_ * KD_ * 64 + 255) / 256), 256, 0, stream>>>(
        slots, dec_pos_w, dec_pos_b, out_dec);
    zero_f32_kernel<<<(int)(((size_t)B_ * EMB_ * RES_ * RES_ + 255) / 256), 256, 0, stream>>>(
        out_loss, (size_t)B_ * EMB_ * RES_ * RES_);
}
